// Involution2D_2989297238556
// MI455X (gfx1250) — matrix operand order repaired and checked by reading, not compiled, not run on hardware
//
#include <hip/hip_runtime.h>
#include <hip/hip_bf16.h>

typedef __attribute__((ext_vector_type(16))) __bf16 v16bf;
typedef __attribute__((ext_vector_type(8)))  __bf16 v8bf;
typedef __attribute__((ext_vector_type(8)))  float  v8f;
typedef __attribute__((ext_vector_type(4)))  float  f4;

#define NPIX 65536          // 4*128*128
#define HDIM 128
#define WDIM 128
#define CIN  256
#define CR   64
#define KKG  144
#define FOUT 256

// -------- workspace layout (bytes), bf16 intermediates --------
#define XI_OFF   0L                                   // xi  : NPIX*256 bf16 = 32 MB
#define W_OFF    (XI_OFF + (long)NPIX*FOUT*2)         // w   : NPIX*144 bf16 = 18 MB
#define WIB_OFF  (W_OFF  + (long)NPIX*KKG*2)          // packed Wi bf16 128 KB
#define WRB_OFF  (WIB_OFF + (long)CIN*FOUT*2)         // packed Wr bf16  32 KB
#define WSB_OFF  (WRB_OFF + (long)CIN*CR*2)           // packed Ws bf16  18 KB
// NOTE: WIB/WRB/WSB are contiguous -> staged to LDS with one linear async copy

// -------- LDS layout inside dynamic smem --------
#define LDS_WI       0
#define LDS_WR       (CIN*FOUT*2)                     // 131072
#define LDS_WS       (LDS_WR + CIN*CR*2)              // 163840
#define LDS_WEIGHTS  (LDS_WS + CR*KKG*2)              // 182272 bytes staged
#define LDS_R        LDS_WEIGHTS                      // 8 waves * 16*64 bf16
#define LDS_TOTAL    (LDS_R + 8*16*64*2)              // 198656 bytes

union AV { v16bf v; v8bf h[2]; };

__device__ __forceinline__ v8f wmma_bf16(v16bf a, v16bf b, v8f c) {
  return __builtin_amdgcn_wmma_f32_16x16x32_bf16(
      /*neg_a=*/false, a, /*neg_b=*/false, b,
      /*c_mod=*/(short)0, c, /*reuse_a=*/false, /*reuse_b=*/false);
}

// Pack f32 row-major weight [K,N] into bf16 WMMA-B fragment layout:
// chunk kc = k/32, tile nt = n/16, lane = ((k%32)/16)*16 + n%16, elem e = k%16
__global__ void pack_w_bf16(const float* __restrict__ W, __bf16* __restrict__ out,
                            int K, int N) {
  int idx = blockIdx.x * blockDim.x + threadIdx.x;
  if (idx >= K * N) return;
  int k = idx / N, n = idx % N;
  int kc = k >> 5, krem = k & 31;
  int lane = ((krem >> 4) << 4) + (n & 15);
  int e = krem & 15;
  int nt = n >> 4, ntiles = N >> 4;
  out[((((long)kc * ntiles + nt) * 32 + lane) << 4) + e] = (__bf16)W[idx];
}

// 8 waves per block, one 16-pixel tile per wave; all packed weights staged in
// LDS once per block via CDNA5 async global->LDS copies (ASYNCcnt path).
__global__ __launch_bounds__(256) void invol_pointwise(
    const float* __restrict__ x,
    const float* __restrict__ br,   const float* __restrict__ gamma,
    const float* __restrict__ beta, const float* __restrict__ mean,
    const float* __restrict__ var,  const float* __restrict__ bs,
    const float* __restrict__ bi,
    const __bf16* __restrict__ Wpacked,   // contiguous WiB|WrB|WsB
    __bf16* __restrict__ xi, __bf16* __restrict__ wdyn) {
  extern __shared__ __align__(32) char smem[];

  const int tid   = threadIdx.x;
  const int lane  = tid & 31;
  const int wave  = tid >> 5;
  const int nn    = lane & 15;
  const int khalf = lane >> 4;

  const long pixbase = ((long)blockIdx.x * 8 + wave) * 16;

  // ---- A fragments: issue the x-tile loads FIRST so they overlap the async
  // weight staging below. A k order used here (contiguous): lanes 0-15 hold K = kc*32+0..15;
  // lanes 16-31 hold K = kc*32+16..31.
  f4 q[8][4];
  const float* xrow = x + (pixbase + nn) * CIN;
  #pragma unroll
  for (int kc = 0; kc < 8; ++kc) {
    const float* p0 = xrow + kc * 32 + khalf * 16;       // elems 0..7
    const float* p1 = p0 + 8;                            // elems 8..15
    q[kc][0] = ((const f4*)p0)[0];
    q[kc][1] = ((const f4*)p0)[1];
    q[kc][2] = ((const f4*)p1)[0];
    q[kc][3] = ((const f4*)p1)[1];
  }

  // ---- stage packed weights (182272 B) global -> LDS, async-tensor path ----
  {
    const char* gsrc = (const char*)Wpacked;
    unsigned ldsbase = (unsigned)(unsigned long long)(&smem[0]);
    for (int i = tid * 16; i < LDS_WEIGHTS; i += 256 * 16) {
      unsigned long long ga = (unsigned long long)(gsrc + i);
      unsigned la = ldsbase + (unsigned)i;
      asm volatile("global_load_async_to_lds_b128 %0, %1, off"
                   :: "v"(la), "v"(ga) : "memory");
    }
    asm volatile("s_wait_asynccnt 0x0" ::: "memory");
  }
  __syncthreads();

  const __bf16* WiL = (const __bf16*)(smem + LDS_WI);
  const __bf16* WrL = (const __bf16*)(smem + LDS_WR);
  const __bf16* WsL = (const __bf16*)(smem + LDS_WS);
  __bf16* rlds = (__bf16*)(smem + LDS_R + wave * (16 * 64 * 2));

  // convert the x tile to bf16 A fragments
  AV a[8];
  #pragma unroll
  for (int kc = 0; kc < 8; ++kc) {
    v8bf h0, h1;
    #pragma unroll
    for (int j = 0; j < 4; ++j) {
      h0[j]     = (__bf16)q[kc][0][j];
      h0[4 + j] = (__bf16)q[kc][1][j];
      h1[j]     = (__bf16)q[kc][2][j];
      h1[4 + j] = (__bf16)q[kc][3][j];
    }
    a[kc].h[0] = h0;
    a[kc].h[1] = h1;
  }

  // ---- GEMM1: r = relu(BN(x @ Wr + br))  [16 x 64] ----
  // Preload all B fragments into distinct registers, then run the WMMA chain
  // back-to-back (no per-WMMA dscnt stall, no WAR-hazard NOPs).
  #pragma unroll
  for (int nt = 0; nt < 4; ++nt) {
    AV bv[8];
    #pragma unroll
    for (int kc = 0; kc < 8; ++kc)
      bv[kc].v = *(const v16bf*)(WrL + (((kc * 4 + nt) * 32 + lane) << 4));
    v8f acc = {};
    #pragma unroll
    for (int kc = 0; kc < 8; ++kc)
      acc = wmma_bf16(a[kc].v, bv[kc].v, acc);
    int d = nt * 16 + nn;
    float s   = gamma[d] * rsqrtf(var[d] + 1e-3f);
    float off = (br[d] - mean[d]) * s + beta[d];
    #pragma unroll
    for (int e = 0; e < 8; ++e) {
      float v = acc[e] * s + off;
      v = v > 0.f ? v : 0.f;
      int m = e + khalf * 8;          // D layout: M = e + 8*(lane/16), N = lane%16
      rlds[m * 64 + d] = (__bf16)v;
    }
  }
  __syncthreads();

  // ---- re-swizzle r: D layout -> A fragment layout (K = 64 -> 2 chunks) ----
  AV ra[2];
  #pragma unroll
  for (int kc = 0; kc < 2; ++kc) {
    const __bf16* p0 = &rlds[nn * 64 + kc * 32 + khalf * 16];
    ra[kc].h[0] = *(const v8bf*)p0;
    ra[kc].h[1] = *(const v8bf*)(p0 + 8);
  }

  // ---- GEMM2: w = r @ Ws + bs  [16 x 144] ----
  #pragma unroll
  for (int nt = 0; nt < 9; ++nt) {
    AV bv[2];
    #pragma unroll
    for (int kc = 0; kc < 2; ++kc)
      bv[kc].v = *(const v16bf*)(WsL + (((kc * 9 + nt) * 32 + lane) << 4));
    v8f acc = {};
    #pragma unroll
    for (int kc = 0; kc < 2; ++kc)
      acc = wmma_bf16(ra[kc].v, bv[kc].v, acc);
    int col = nt * 16 + nn;
    float bias = bs[col];
    #pragma unroll
    for (int e = 0; e < 8; ++e) {
      int m = e + khalf * 8;
      wdyn[(pixbase + m) * KKG + col] = (__bf16)(acc[e] + bias);
    }
  }

  // ---- GEMM3: xi = x @ Wi + bi  [16 x 256] ----
  for (int nt = 0; nt < 16; ++nt) {
    AV bv[8];
    #pragma unroll
    for (int kc = 0; kc < 8; ++kc)
      bv[kc].v = *(const v16bf*)(WiL + (((kc * 16 + nt) * 32 + lane) << 4));
    v8f acc = {};
    #pragma unroll
    for (int kc = 0; kc < 8; ++kc)
      acc = wmma_bf16(a[kc].v, bv[kc].v, acc);
    int col = nt * 16 + nn;
    float bias = bi[col];
    #pragma unroll
    for (int e = 0; e < 8; ++e) {
      int m = e + khalf * 8;
      xi[(pixbase + m) * FOUT + col] = (__bf16)(acc[e] + bias);
    }
  }
}

// 3x3 dynamic aggregation with the faithful flat-index remap:
// n = g*144 + fg*9 + kk2 ; kk = n/256 ; c = n%256
__global__ __launch_bounds__(256) void invol_aggregate(
    const __bf16* __restrict__ xi, const __bf16* __restrict__ wdyn,
    float* __restrict__ out) {
  const int pix = blockIdx.x;
  const int ch  = threadIdx.x;
  const int g = ch >> 4, fg = ch & 15;
  const int w0 = pix & 127;
  const int h0 = (pix >> 7) & 127;
  const int b  = pix >> 14;
  const __bf16* wrow = wdyn + (long)pix * KKG + g * 9;
  float wv[9];
  #pragma unroll
  for (int kk2 = 0; kk2 < 9; ++kk2) wv[kk2] = (float)wrow[kk2];
  float sum = 0.f;
  #pragma unroll
  for (int kk2 = 0; kk2 < 9; ++kk2) {
    int n  = g * 144 + fg * 9 + kk2;
    int kk = n >> 8;
    int c  = n & 255;
    int ki = kk / 3, kj = kk - ki * 3;
    int hh = h0 + ki - 1, ww = w0 + kj - 1;
    if ((unsigned)hh < 128u && (unsigned)ww < 128u) {
      long q = ((((long)b * HDIM + hh) * WDIM) + ww) * FOUT + c;
      sum += wv[kk2] * (float)xi[q];
    }
  }
  out[(long)pix * FOUT + ch] = sum;
}

extern "C" void kernel_launch(void* const* d_in, const int* in_sizes, int n_in,
                              void* d_out, int out_size, void* d_ws, size_t ws_size,
                              hipStream_t stream) {
  (void)in_sizes; (void)n_in; (void)out_size; (void)ws_size;
  const float* x     = (const float*)d_in[0];
  const float* Wr    = (const float*)d_in[1];
  const float* br    = (const float*)d_in[2];
  const float* gamma = (const float*)d_in[3];
  const float* beta  = (const float*)d_in[4];
  const float* mean  = (const float*)d_in[5];
  const float* var   = (const float*)d_in[6];
  const float* Ws    = (const float*)d_in[7];
  const float* bs    = (const float*)d_in[8];
  const float* Wi    = (const float*)d_in[9];
  const float* bi    = (const float*)d_in[10];
  float* out = (float*)d_out;

  char* ws = (char*)d_ws;
  __bf16* xi   = (__bf16*)(ws + XI_OFF);
  __bf16* wdyn = (__bf16*)(ws + W_OFF);
  __bf16* WiB  = (__bf16*)(ws + WIB_OFF);
  __bf16* WrB  = (__bf16*)(ws + WRB_OFF);
  __bf16* WsB  = (__bf16*)(ws + WSB_OFF);

  (void)hipFuncSetAttribute((const void*)invol_pointwise,
                            hipFuncAttributeMaxDynamicSharedMemorySize,
                            LDS_TOTAL);

  pack_w_bf16<<<(CIN * FOUT + 255) / 256, 256, 0, stream>>>(Wi, WiB, CIN, FOUT);
  pack_w_bf16<<<(CIN * CR   + 255) / 256, 256, 0, stream>>>(Wr, WrB, CIN, CR);
  pack_w_bf16<<<(CR * KKG   + 255) / 256, 256, 0, stream>>>(Ws, WsB, CR, KKG);

  invol_pointwise<<<NPIX / (16 * 8), 256, LDS_TOTAL, stream>>>(
      x, br, gamma, beta, mean, var, bs, bi, WiB, xi, wdyn);

  invol_aggregate<<<NPIX, 256, 0, stream>>>(xi, wdyn, out);
}
